// AMPBlock0_17291538334027
// MI455X (gfx1250) — matrix operand order repaired and checked by reading, not compiled, not run on hardware
//
#include <hip/hip_runtime.h>

// ---------------------------------------------------------------------------
// AMP residual block (BigVGAN-style) for MI455X / gfx1250.
//   B=8, C=512, T=8192, fp32 in/out.
//   conv1/conv2 (512x512x3) run as bf16 WMMA GEMMs (f32 accumulate).
//   Activation outputs stored directly as bf16 (precision-neutral: they feed
//   only the bf16 GEMM inputs) -> halves intermediate HBM traffic.
// ---------------------------------------------------------------------------

#define CH   512
#define NB   8
#define TIN0 8192

typedef __attribute__((ext_vector_type(16))) __bf16 v16bf;
typedef __attribute__((ext_vector_type(8)))  __bf16 v8bf;
typedef __attribute__((ext_vector_type(8)))  float  v8f;

static __device__ __forceinline__ unsigned short f32_to_bf16_raw(float f) {
    unsigned u = __float_as_uint(f);
    u += 0x7FFFu + ((u >> 16) & 1u);          // round-to-nearest-even
    return (unsigned short)(u >> 16);
}

// B fragment: 16 consecutive bf16 (two b128 LDS reads)
static __device__ __forceinline__ v16bf ld_frag_b(const unsigned short* p) {
    const v8bf* q = reinterpret_cast<const v8bf*>(p);
    v8bf lo = q[0], hi = q[1];
    v16bf r;
#pragma unroll
    for (int i = 0; i < 8; ++i) { r[i] = lo[i]; r[i + 8] = hi[i]; }
    return r;
}

// A fragment: 16 consecutive K elements (two b128 reads)
static __device__ __forceinline__ v16bf ld_frag_a(const unsigned short* p) {
    const v8bf* q = reinterpret_cast<const v8bf*>(p);
    v8bf lo = q[0], hi = q[1];
    v16bf r;
#pragma unroll
    for (int i = 0; i < 8; ++i) { r[i] = lo[i]; r[i + 8] = hi[i]; }
    return r;
}

// ---------------------------------------------------------------------------
// Kernel 1: weight-norm prep.  w[o,c,k] = g[o]*v[o,c,k]/||v[o]||, stored bf16
// in GEMM-A layout A[o][k*512 + c]  (K index = kernel_tap*512 + in_channel).
// grid.x = 1024: first 512 blocks -> (v1,g1)->A1, next 512 -> (v2,g2)->A2.
// ---------------------------------------------------------------------------
__global__ void prep_weights_kernel(const float* __restrict__ v1,
                                    const float* __restrict__ g1,
                                    const float* __restrict__ v2,
                                    const float* __restrict__ g2,
                                    unsigned short* __restrict__ A1,
                                    unsigned short* __restrict__ A2) {
    __shared__ float red[256];
    const int wid = blockIdx.x;
    const int set = wid >> 9;          // 0 or 1
    const int o   = wid & 511;
    const int tid = threadIdx.x;

    const float* v = (set == 0) ? v1 : v2;
    const float* g = (set == 0) ? g1 : g2;
    unsigned short* A = (set == 0) ? A1 : A2;

    const float* vo = v + (size_t)o * (CH * 3);

    float part = 0.f;
    for (int i = tid; i < CH * 3; i += 256) {
        float x = vo[i];
        part += x * x;
    }
    red[tid] = part;
    __syncthreads();
    for (int s = 128; s > 0; s >>= 1) {
        if (tid < s) red[tid] += red[tid + s];
        __syncthreads();
    }
    const float scale = g[o] * __frsqrt_rn(red[0]);

    unsigned short* Ao = A + (size_t)o * (CH * 3);
    for (int i = tid; i < CH * 3; i += 256) {
        int c = i / 3, kp = i % 3;
        Ao[kp * CH + c] = f32_to_bf16_raw(scale * vo[i]);
    }
}

// ---------------------------------------------------------------------------
// Kernel 2: fused anti-aliased activation:
//   upsample x2 (2-phase 6-tap kaiser-sinc) -> snake_beta -> downsample x2
//   (12-tap, stride 2).  One output element per thread.  fp32 compute,
//   bf16 store (output feeds the bf16 WMMA GEMM only).
// ---------------------------------------------------------------------------
__global__ void act_kernel(const float* __restrict__ in, int Tin,
                           const float* __restrict__ alpha,
                           const float* __restrict__ beta,
                           const float* __restrict__ upf,   // (2C,1,6) flat
                           const float* __restrict__ dnf,   // (C,1,12) flat
                           unsigned short* __restrict__ out, int Tact) {
    long gid = (long)blockIdx.x * blockDim.x + threadIdx.x;
    long total = (long)NB * CH * Tact;
    if (gid >= total) return;

    int t = (int)(gid % Tact);
    int c = (int)((gid / Tact) % CH);
    int b = (int)(gid / ((long)Tact * CH));

    const float a   = __expf(alpha[c]);
    const float den = 2.f * __expf(beta[c]) + 1e-9f;
    const float* xin = in + ((long)b * CH + c) * Tin;
    const float* up0 = upf + (2 * c) * 6;
    const float* up1 = up0 + 6;
    const float* f   = dnf + c * 12;

    const int Lup = 2 * Tin - 4;
    float acc = 0.f;
#pragma unroll
    for (int k = 0; k < 12; ++k) {
        int u = 2 * t + k - 5;                 // downsample: stride 2, pad 5
        if (u < 0 || u >= Lup) continue;
        int vv = u + 2;                        // undo the [2:-2] crop
        int tu = vv >> 1;
        const float* ph = (vv & 1) ? up1 : up0;
        float s = 0.f;
#pragma unroll
        for (int j = 0; j < 6; ++j) {          // polyphase upsample tap
            int xi = tu + j - 2;
            if (xi >= 0 && xi < Tin) s += ph[j] * xin[xi];
        }
        float sn = s + (1.f - __cosf(2.f * a * s)) / den;   // snake-beta
        acc += f[k] * sn;
    }
    out[((long)b * CH + c) * Tact + t] = f32_to_bf16_raw(acc);
}

// ---------------------------------------------------------------------------
// Kernel 3: conv1d(C,C,3,pad=1) as WMMA GEMM.  M=512, K=1536, N=Tout.
// Block = 256 threads (8 wave32).  Each block: all 512 out-channels x 32 t.
// Wave w owns rows [64w, 64w+64) x 32 cols -> 8 accumulators of 16x16.
// Input (bf16) tile (512 ch x 34 t) staged in LDS, layout [t][c], stride 520.
// K loop split kpos(3) x cb(16) so all ds/global offsets become immediates.
// ---------------------------------------------------------------------------
#define LDSW 520                                  // padded row stride (elems)
#define NT   32                                   // time columns per block

__global__ void __launch_bounds__(256)
gemm_conv3_kernel(const unsigned short* __restrict__ in, int Tin,  // bf16
                  const unsigned short* __restrict__ Abf,   // [512][1536] bf16
                  const float* __restrict__ bias,
                  float* __restrict__ out, int N,
                  const float* __restrict__ residual, int Tres) {
    __shared__ unsigned short xs[(NT + 2) * LDSW];

    const int tid  = threadIdx.x;
    const int lane = tid & 31;
    const int wv   = tid >> 5;                    // wave id 0..7
    const int t0   = blockIdx.x * NT;
    const int b    = blockIdx.y;

    // ---- stage input tile: channels 0..511, times t0-1 .. t0+NT (conv pad 1)
    {
        const int c0 = tid * 2;
#pragma unroll
        for (int cc = 0; cc < 2; ++cc) {
            const int c = c0 + cc;
            const unsigned short* src = in + ((long)b * CH + c) * Tin;
            for (int tt = 0; tt < NT + 2; ++tt) {
                int t = t0 - 1 + tt;
                xs[tt * LDSW + c] =
                    (t >= 0 && t < Tin) ? src[t] : (unsigned short)0;
            }
        }
    }
    __syncthreads();

    const int col    = lane & 15;                 // D column (time) in tile
    const int r16    = lane & 15;                 // A row within 16
    const int hi     = lane >> 4;                 // lane half
    const int khalf  = hi << 4;                   // A: K-run base 0/16
    const int khalf2 = hi << 4;                   // B: K-run base 0/16
    const int mwave  = wv * 64;

    v8f acc[8];
#pragma unroll
    for (int s = 0; s < 8; ++s)
        acc[s] = (v8f){0.f,0.f,0.f,0.f,0.f,0.f,0.f,0.f};

    // A base for this lane's row; s-subtile / cb offsets are immediates.
    const unsigned short* Abase =
        Abf + (size_t)(mwave + r16) * (CH * 3) + khalf;

    // ---- K loop: K index = kpos*512 + c ; 3 taps x 16 channel-blocks of 32
#pragma unroll
    for (int kpos = 0; kpos < 3; ++kpos) {
        const unsigned short* bp0 = &xs[(col + kpos) * LDSW + khalf2];
        const unsigned short* bp1 = &xs[(col + 16 + kpos) * LDSW + khalf2];
        const unsigned short* ap  = Abase + kpos * CH;
#pragma unroll 4
        for (int cb = 0; cb < 16; ++cb) {
            const int e = cb * 32;
            // issue all loads for this K-step first (distinct regs),
            // then the 8 WMMAs -> loads drain while matrix ops run
            v16bf b0 = ld_frag_b(bp0 + e);
            v16bf b1 = ld_frag_b(bp1 + e);
            v16bf a0 = ld_frag_a(ap + e);
            v16bf a1 = ld_frag_a(ap + 1 * 16 * CH * 3 + e);
            v16bf a2 = ld_frag_a(ap + 2 * 16 * CH * 3 + e);
            v16bf a3 = ld_frag_a(ap + 3 * 16 * CH * 3 + e);

            acc[0] = __builtin_amdgcn_wmma_f32_16x16x32_bf16(
                false, a0, false, b0, (short)0, acc[0], false, false);
            acc[4] = __builtin_amdgcn_wmma_f32_16x16x32_bf16(
                false, a0, false, b1, (short)0, acc[4], false, false);
            acc[1] = __builtin_amdgcn_wmma_f32_16x16x32_bf16(
                false, a1, false, b0, (short)0, acc[1], false, false);
            acc[5] = __builtin_amdgcn_wmma_f32_16x16x32_bf16(
                false, a1, false, b1, (short)0, acc[5], false, false);
            acc[2] = __builtin_amdgcn_wmma_f32_16x16x32_bf16(
                false, a2, false, b0, (short)0, acc[2], false, false);
            acc[6] = __builtin_amdgcn_wmma_f32_16x16x32_bf16(
                false, a2, false, b1, (short)0, acc[6], false, false);
            acc[3] = __builtin_amdgcn_wmma_f32_16x16x32_bf16(
                false, a3, false, b0, (short)0, acc[3], false, false);
            acc[7] = __builtin_amdgcn_wmma_f32_16x16x32_bf16(
                false, a3, false, b1, (short)0, acc[7], false, false);
        }
    }

    // ---- epilogue: bias (+ residual) and store, two 16-column groups
#pragma unroll
    for (int g = 0; g < 2; ++g) {
        const int tcol = t0 + col + g * 16;
        if (tcol < N) {
#pragma unroll
            for (int s = 0; s < 4; ++s) {
#pragma unroll
                for (int r = 0; r < 8; ++r) {
                    int m = mwave + s * 16 + hi * 8 + r;
                    float v = acc[g * 4 + s][r] + bias[m];
                    if (residual)
                        v += residual[((long)b * CH + m) * Tres + tcol];
                    out[((long)b * CH + m) * (long)N + tcol] = v;
                }
            }
        }
    }
}

// ---------------------------------------------------------------------------
// Launch: prep weights -> act1 -> gemm1(+b1) -> act2 -> gemm2(+b2+residual)
// ---------------------------------------------------------------------------
extern "C" void kernel_launch(void* const* d_in, const int* in_sizes, int n_in,
                              void* d_out, int out_size, void* d_ws, size_t ws_size,
                              hipStream_t stream) {
    const float* x      = (const float*)d_in[0];
    const float* alpha1 = (const float*)d_in[1];
    const float* beta1  = (const float*)d_in[2];
    const float* alpha2 = (const float*)d_in[3];
    const float* beta2  = (const float*)d_in[4];
    const float* v1     = (const float*)d_in[5];
    const float* g1     = (const float*)d_in[6];
    const float* b1     = (const float*)d_in[7];
    const float* v2     = (const float*)d_in[8];
    const float* g2     = (const float*)d_in[9];
    const float* b2     = (const float*)d_in[10];
    const float* upf    = (const float*)d_in[11];
    const float* dnf    = (const float*)d_in[12];
    float* out = (float*)d_out;

    const int T1 = TIN0 - 2;   // 8190 after act1 / conv1
    const int T2 = TIN0 - 4;   // 8188 after act2 / conv2 (== output length)

    // workspace layout
    char* ws = (char*)d_ws;
    const size_t Abytes = (size_t)CH * CH * 3 * sizeof(unsigned short); // 1.5MB
    unsigned short* A1 = (unsigned short*)(ws);
    unsigned short* A2 = (unsigned short*)(ws + Abytes);
    // act outputs (bf16), conv1 output (f32)
    unsigned short* bufA = (unsigned short*)(ws + 2 * Abytes);
    float* bufB = (float*)(ws + 2 * Abytes + (size_t)NB * CH * T1 * 2);

    // 1) weight-norm -> bf16 A matrices
    prep_weights_kernel<<<dim3(1024), dim3(256), 0, stream>>>(
        v1, g1, v2, g2, A1, A2);

    // 2) activation1d #1 : x (f32, T=8192) -> bufA (bf16, T1=8190)
    {
        long total = (long)NB * CH * T1;
        int blocks = (int)((total + 255) / 256);
        act_kernel<<<dim3(blocks), dim3(256), 0, stream>>>(
            x, TIN0, alpha1, beta1, upf, dnf, bufA, T1);
    }

    // 3) conv1 (WMMA GEMM) : bufA (bf16) -> bufB (f32), + b1
    {
        int ntiles = (T1 + NT - 1) / NT;   // 256
        gemm_conv3_kernel<<<dim3(ntiles, NB), dim3(256), 0, stream>>>(
            bufA, T1, A1, b1, bufB, T1, nullptr, 0);
    }

    // 4) activation1d #2 : bufB (f32, T1) -> bufA (bf16, T2=8188)
    {
        long total = (long)NB * CH * T2;
        int blocks = (int)((total + 255) / 256);
        act_kernel<<<dim3(blocks), dim3(256), 0, stream>>>(
            bufB, T1, alpha2, beta2, upf, dnf, bufA, T2);
    }

    // 5) conv2 (WMMA GEMM) : bufA (bf16) -> d_out (f32), + b2 + residual
    {
        int ntiles = (T2 + NT - 1) / NT;   // 256
        gemm_conv3_kernel<<<dim3(ntiles, NB), dim3(256), 0, stream>>>(
            bufA, T2, A2, b2, out, T2, x, TIN0);
    }
}
